// LTAE_19765439496417
// MI455X (gfx1250) — matrix operand order repaired and checked by reading, not compiled, not run on hardware
//
#include <hip/hip_runtime.h>
#include <hip/hip_bf16.h>

// ---------------------------------------------------------------------------
// LTAE forward for MI455X (gfx1250). One 256-thread block (8 wave32) per
// spatial pixel (BS*H*W = 2048 blocks). All matmuls on v_wmma_f32_16x16x32_f16
// with fp32 accumulate; f16 weight copies are produced into d_ws each launch.
// ---------------------------------------------------------------------------

typedef __attribute__((ext_vector_type(16))) _Float16 v16h;
typedef __attribute__((ext_vector_type(8)))  _Float16 v8h;
typedef __attribute__((ext_vector_type(8)))  float    v8f;

#define S_LEN 30
#define NCH   128     // NC
#define NHEAD 16
#define MLPH  256

// d_ws layout in _Float16 elements
#define WS_FC1 0              // 128*128
#define WS_FC2 16384          // 128*2048
#define WS_M1  278528         // 256*128
#define WS_M2  311296         // 128*256
#define WS_TOT 344064

// --------------------------- WMMA helpers ----------------------------------

__device__ __forceinline__ v8f wmma16(v16h a, v16h b, v8f c) {
  // v_wmma_f32_16x16x32_f16  (neg_a, A, neg_b, B, c_mod, C, reuse_a, reuse_b)
  return __builtin_amdgcn_wmma_f32_16x16x32_f16(false, a, false, b,
                                                (short)0, c, false, false);
}

// A-matrix (16x32 f16) load from row-major f16 [m][k] with leading dim ld.
// Layout: lane m = lane&15; lanes 0-15: K = 0..15;
// lanes 16-31: K = 16..31 (contiguous, matches loadB_h).
__device__ __forceinline__ v16h loadA_h(const _Float16* base, int m0, int ld,
                                        int k0, int lane) {
  int m  = m0 + (lane & 15);
  int hi = lane >> 4;
  const _Float16* p = base + m * ld + k0 + hi * 16;
  union { v16h v; v8h h[2]; } u;
  u.h[0] = *(const v8h*)p;
  u.h[1] = *(const v8h*)(p + 8);
  return u.v;
}

// B-matrix (32x16 f16) for B[k][n] = W[n][k], W row-major [n][k], ld halfs.
// Layout: lane column n = n0 + (lane&15); lanes 0-15 hold K=k0..k0+15,
// lanes 16-31 hold K=k0+16..k0+31 (contiguous halfwords).
__device__ __forceinline__ v16h loadB_h(const _Float16* base, int n0, int ld,
                                        int k0, int lane) {
  int n  = n0 + (lane & 15);
  int hi = lane >> 4;
  return *(const v16h*)(base + (size_t)n * ld + k0 + hi * 16);
}

__device__ __forceinline__ float gelu_exact(float x) {
  return 0.5f * x * (1.f + erff(x * 0.70710678118654752f));
}

// --------------------------- weight fp32 -> fp16 ---------------------------

__global__ __launch_bounds__(256) void cvt_weights(
    const float* __restrict__ fc1, const float* __restrict__ fc2,
    const float* __restrict__ m1,  const float* __restrict__ m2,
    _Float16* __restrict__ ws) {
  int i = blockIdx.x * 256 + threadIdx.x;
  if (i >= WS_TOT) return;
  if (i < WS_FC2)      ws[i] = (_Float16)fc1[i - WS_FC1];
  else if (i < WS_M1)  ws[i] = (_Float16)fc2[i - WS_FC2];
  else if (i < WS_M2)  ws[i] = (_Float16)m1[i - WS_M1];
  else                 ws[i] = (_Float16)m2[i - WS_M2];
}

// ------------------------------ main kernel --------------------------------

__global__ __launch_bounds__(256) void ltae_kernel(
    const float* __restrict__ x, const int* __restrict__ doy,
    const unsigned char* __restrict__ pmask,
    const float* __restrict__ fc1_b, const float* __restrict__ fc2_b,
    const float* __restrict__ m1_b,  const float* __restrict__ m2_b,
    const float* __restrict__ gn1_w, const float* __restrict__ gn1_b,
    const float* __restrict__ gn2_w, const float* __restrict__ gn2_b,
    const float* __restrict__ gn3_w, const float* __restrict__ gn3_b,
    const _Float16* __restrict__ wsh, float* __restrict__ out) {
  // LDS: 36384 floats = 145.5 KB (within 320KB/WGP)
  __shared__ float smem[36384];
  float*     sV     = smem;                            // [30][128] v ; later gn2 output
  _Float16*  sVh    = (_Float16*)(smem + 3840);        // [32][128] f16 (pad rows 0)
  float*     sKQ    = smem + 5888;                     // [30][128] kq ; later fc2 out / final
  float*     sAttn  = smem + 9728;                     // [16][32][32] f32 scores
  _Float16*  sM1h   = (_Float16*)(smem + 9728);        // [32][256] f16 mlp hidden (alias)
  _Float16*  sAttnH = (_Float16*)(smem + 26112);       // [16][32][32] f16 probs
  _Float16*  sTmp   = (_Float16*)(smem + 34304);       // [8 waves][16 o][32 t] f16
  float*     sRed   = smem + 36352;                    // [32] groupnorm stats

  const int tid  = threadIdx.x;
  const int lane = tid & 31;
  const int w    = tid >> 5;             // wave id 0..7
  const int p    = blockIdx.x;           // pixel 0..2047
  const int b    = p >> 10;              // batch
  const int hpix = p & 1023;             // hh*32+ww

  const _Float16* fc1h = wsh + WS_FC1;
  const _Float16* fc2h = wsh + WS_FC2;
  const _Float16* m1h  = wsh + WS_M1;
  const _Float16* m2h  = wsh + WS_M2;

  // ---- Phase 0: load x slice, groupnorm gn1, add positional encoding ------
  for (int i = tid; i < S_LEN * NCH; i += 256) {
    int t = i >> 7, c = i & 127;
    sV[i] = x[(size_t)((b * S_LEN + t) * NCH + c) * 1024 + hpix];
  }
  __syncthreads();
  if (tid < 16) {
    float s = 0.f, ss = 0.f;
    for (int c8 = 0; c8 < 8; ++c8)
      for (int t = 0; t < S_LEN; ++t) {
        float v = sV[t * NCH + tid * 8 + c8];
        s += v; ss += v * v;
      }
    float mu  = s * (1.f / 240.f);
    float var = ss * (1.f / 240.f) - mu * mu;
    sRed[2 * tid]     = mu;
    sRed[2 * tid + 1] = rsqrtf(var + 1e-5f);
  }
  __syncthreads();
  for (int i = tid; i < 32 * NCH; i += 256) {
    int t = i >> 7, c = i & 127;
    if (t < S_LEN) {
      int g = c >> 3, j = c & 7;
      float v = (sV[t * NCH + c] - sRed[2 * g]) * sRed[2 * g + 1] * gn1_w[c] + gn1_b[c];
      // denom[j] = 1000^((j+1)/4) ; even j -> sin, odd j -> cos
      float denom = __expf(1.7269388f * (float)(j + 1));
      float ang   = (float)doy[b * S_LEN + t] / denom;
      v += (j & 1) ? cosf(ang) : sinf(ang);
      sV[t * NCH + c] = v;
      sVh[i] = (_Float16)v;
    } else {
      sVh[i] = (_Float16)0.f;           // zero-pad rows 30,31 for WMMA K/M
    }
  }
  __syncthreads();

  // ---- Phase 1: kq = v @ fc1_w^T + fc1_b (WMMA) ---------------------------
  {
    int n0 = w * 16;
    float bias = fc1_b[n0 + (lane & 15)];
    for (int mt = 0; mt < 2; ++mt) {
      v8f acc = {bias, bias, bias, bias, bias, bias, bias, bias};
      for (int kt = 0; kt < 4; ++kt) {
        v16h a  = loadA_h(sVh, mt * 16, NCH, kt * 32, lane);
        v16h bm = loadB_h(fc1h, n0, NCH, kt * 32, lane);
        acc = wmma16(a, bm, acc);
      }
      int n  = n0 + (lane & 15);
      int rb = mt * 16 + ((lane >> 4) << 3);
#pragma unroll
      for (int j = 0; j < 8; ++j) {
        int m = rb + j;
        if (m < S_LEN) sKQ[m * NCH + n] = acc[j];
      }
    }
  }
  __syncthreads();

  // ---- Phase 2: scores (DK=4), mask, softmax; emit attn output ------------
  // zero-pad the f16 prob mirror: cols k=30,31 (all q) and rows q=30,31 (all k)
  for (int i = tid; i < NHEAD * 32 * 2; i += 256) {
    int h = i >> 6, q = (i >> 1) & 31, k = 30 + (i & 1);
    sAttnH[(h * 32 + q) * 32 + k] = (_Float16)0.f;
  }
  for (int i = tid; i < NHEAD * 2 * 32; i += 256) {
    int h = i >> 6, q = 30 + ((i >> 5) & 1), k = i & 31;
    sAttnH[(h * 32 + q) * 32 + k] = (_Float16)0.f;
  }
  for (int i = tid; i < NHEAD * S_LEN * S_LEN; i += 256) {
    int h = i / 900, r = i % 900, q = r / 30, k = r % 30;
    const float* K = sKQ + k * NCH + h * 8;
    const float* Q = sKQ + q * NCH + h * 8 + 4;
    float sc = 0.5f * (Q[0] * K[0] + Q[1] * K[1] + Q[2] * K[2] + Q[3] * K[3]);
    if (pmask[b * S_LEN + k]) sc = -1000.f;
    sAttn[(h * 32 + q) * 32 + k] = sc;
  }
  __syncthreads();
  for (int row = tid; row < NHEAD * S_LEN; row += 256) {
    int h = row / 30, q = row % 30;
    float* r = sAttn + (h * 32 + q) * 32;
    _Float16* rh = sAttnH + (h * 32 + q) * 32;
    float mx = -1e30f;
    for (int k = 0; k < 30; ++k) mx = fmaxf(mx, r[k]);
    float sum = 0.f;
    for (int k = 0; k < 30; ++k) { float e = __expf(r[k] - mx); r[k] = e; sum += e; }
    float inv = 1.f / sum;
    size_t abase = 7864320ull +
                   ((size_t)((b * NHEAD + h) * S_LEN + q) * S_LEN) * 1024ull +
                   (size_t)hpix;
    for (int k = 0; k < 30; ++k) {
      float pr = r[k] * inv;
      rh[k] = (_Float16)pr;
      out[abase + (size_t)k * 1024] = pr;   // attn output (BS,NH,S,S,H,W)
    }
  }
  __syncthreads();

  // ---- Phase 3: fc2out = sum_h attn_h @ (v @ W2_h^T), + bias + v ----------
  {
    int n0 = w * 16;                          // this wave's fc2 output columns
    _Float16* myT = sTmp + w * (16 * 32);     // [o_local][t] f16 scratch
    // hoist the loop-invariant A tiles of v (16x32 each): 8 x v16h in VGPRs
    v16h aV[8];
#pragma unroll
    for (int mt = 0; mt < 2; ++mt)
#pragma unroll
      for (int kt = 0; kt < 4; ++kt)
        aV[mt * 4 + kt] = loadA_h(sVh, mt * 16, NCH, kt * 32, lane);

    v8f acc0 = {0.f, 0.f, 0.f, 0.f, 0.f, 0.f, 0.f, 0.f};
    v8f acc1 = {0.f, 0.f, 0.f, 0.f, 0.f, 0.f, 0.f, 0.f};
    for (int h = 0; h < NHEAD; ++h) {
      // VW2_h[t][o] = sum_c v[t][c] * fc2_w[o][h*128+c]
#pragma unroll
      for (int mt = 0; mt < 2; ++mt) {
        v8f d = {0.f, 0.f, 0.f, 0.f, 0.f, 0.f, 0.f, 0.f};
#pragma unroll
        for (int kt = 0; kt < 4; ++kt) {
          v16h bm = loadB_h(fc2h, n0, NHEAD * NCH, h * NCH + kt * 32, lane);
          d = wmma16(aV[mt * 4 + kt], bm, d);
        }
        // transpose-store D (rows t) into [o][t] so it can be re-read in B layout
        int o = lane & 15, hi = lane >> 4;
        union { v8h v; _Float16 e[8]; } u;
#pragma unroll
        for (int j = 0; j < 8; ++j) u.e[j] = (_Float16)d[j];
        *(v8h*)(myT + o * 32 + mt * 16 + hi * 8) = u.v;
      }
      // B2[k=t][n=o] from myT ; A = attn_h (q x t) from f16 prob mirror
      v16h b2 = *(const v16h*)(myT + (lane & 15) * 32 + (lane >> 4) * 16);
      v16h a0 = loadA_h(sAttnH + h * 1024, 0,  32, 0, lane);
      v16h a1 = loadA_h(sAttnH + h * 1024, 16, 32, 0, lane);
      acc0 = wmma16(a0, b2, acc0);
      acc1 = wmma16(a1, b2, acc1);
    }
    float bias = fc2_b[n0 + (lane & 15)];
    float* sOut = sKQ;                        // reuse (kq consumed in phase 2)
    int n = n0 + (lane & 15);
    int rb = (lane >> 4) << 3;
#pragma unroll
    for (int j = 0; j < 8; ++j) {
      int m = rb + j;
      if (m < S_LEN) sOut[m * NCH + n] = acc0[j] + bias + sV[m * NCH + n];
    }
    rb = 16 + ((lane >> 4) << 3);
#pragma unroll
    for (int j = 0; j < 8; ++j) {
      int m = rb + j;
      if (m < S_LEN) sOut[m * NCH + n] = acc1[j] + bias + sV[m * NCH + n];
    }
  }
  __syncthreads();

  // ---- Phase 4: groupnorm gn2 -> sOutN (f32, reuse sV) + f16 copy ---------
  float* sOut  = sKQ;
  float* sOutN = sV;
  _Float16* sOutNh = sVh;
  if (tid < 16) {
    float s = 0.f, ss = 0.f;
    for (int c8 = 0; c8 < 8; ++c8)
      for (int t = 0; t < S_LEN; ++t) {
        float v = sOut[t * NCH + tid * 8 + c8];
        s += v; ss += v * v;
      }
    float mu  = s * (1.f / 240.f);
    float var = ss * (1.f / 240.f) - mu * mu;
    sRed[2 * tid]     = mu;
    sRed[2 * tid + 1] = rsqrtf(var + 1e-5f);
  }
  __syncthreads();
  for (int i = tid; i < 32 * NCH; i += 256) {
    int t = i >> 7, c = i & 127;
    if (t < S_LEN) {
      int g = c >> 3;
      float v = (sOut[t * NCH + c] - sRed[2 * g]) * sRed[2 * g + 1] * gn2_w[c] + gn2_b[c];
      sOutN[t * NCH + c] = v;
      sOutNh[i] = (_Float16)v;
    } else {
      sOutNh[i] = (_Float16)0.f;
    }
  }
  __syncthreads();

  // ---- Phase 5: mlp hidden = gelu(out @ mlp1_w^T + b) -> sM1h f16 ---------
  for (int nt = w; nt < 16; nt += 8) {
    int n0 = nt * 16;
    float bias = m1_b[n0 + (lane & 15)];
    for (int mt = 0; mt < 2; ++mt) {
      v8f d = {bias, bias, bias, bias, bias, bias, bias, bias};
      for (int kt = 0; kt < 4; ++kt) {
        v16h a  = loadA_h(sOutNh, mt * 16, NCH, kt * 32, lane);
        v16h bm = loadB_h(m1h, n0, NCH, kt * 32, lane);
        d = wmma16(a, bm, d);
      }
      int n  = n0 + (lane & 15);
      int rb = mt * 16 + ((lane >> 4) << 3);
#pragma unroll
      for (int j = 0; j < 8; ++j)
        sM1h[(rb + j) * MLPH + n] = (_Float16)gelu_exact(d[j]);
    }
  }
  __syncthreads();

  // ---- Phase 6: out = gn2out + gelu(hidden @ mlp2_w^T + b) ----------------
  {
    int n0 = w * 16;
    float bias = m2_b[n0 + (lane & 15)];
    float* sFin = sKQ;                        // reuse again
    for (int mt = 0; mt < 2; ++mt) {
      v8f d = {bias, bias, bias, bias, bias, bias, bias, bias};
      for (int kt = 0; kt < 8; ++kt) {
        v16h a  = loadA_h(sM1h, mt * 16, MLPH, kt * 32, lane);
        v16h bm = loadB_h(m2h, n0, MLPH, kt * 32, lane);
        d = wmma16(a, bm, d);
      }
      int n  = n0 + (lane & 15);
      int rb = mt * 16 + ((lane >> 4) << 3);
#pragma unroll
      for (int j = 0; j < 8; ++j) {
        int m = rb + j;
        if (m < S_LEN)
          sFin[m * NCH + n] = sOutN[m * NCH + n] + gelu_exact(d[j]);
      }
    }
  }
  __syncthreads();

  // ---- Phase 7: groupnorm gn3, scatter to output (BS,S,C,H,W) -------------
  {
    float* sFin = sKQ;
    if (tid < 16) {
      float s = 0.f, ss = 0.f;
      for (int c8 = 0; c8 < 8; ++c8)
        for (int t = 0; t < S_LEN; ++t) {
          float v = sFin[t * NCH + tid * 8 + c8];
          s += v; ss += v * v;
        }
      float mu  = s * (1.f / 240.f);
      float var = ss * (1.f / 240.f) - mu * mu;
      sRed[2 * tid]     = mu;
      sRed[2 * tid + 1] = rsqrtf(var + 1e-5f);
    }
    __syncthreads();
    for (int i = tid; i < S_LEN * NCH; i += 256) {
      int t = i >> 7, c = i & 127, g = c >> 3;
      float v = (sFin[i] - sRed[2 * g]) * sRed[2 * g + 1] * gn3_w[c] + gn3_b[c];
      out[(size_t)((b * S_LEN + t) * NCH + c) * 1024 + hpix] = v;
    }
  }
}

// ------------------------------ launcher -----------------------------------

extern "C" void kernel_launch(void* const* d_in, const int* in_sizes, int n_in,
                              void* d_out, int out_size, void* d_ws, size_t ws_size,
                              hipStream_t stream) {
  (void)in_sizes; (void)n_in; (void)out_size; (void)ws_size;
  const float* x      = (const float*)d_in[0];
  const int*   doy    = (const int*)d_in[1];
  const unsigned char* pmask = (const unsigned char*)d_in[2];  // jnp bool = u8
  const float* fc1_w  = (const float*)d_in[3];
  const float* fc1_b  = (const float*)d_in[4];
  const float* fc2_w  = (const float*)d_in[5];
  const float* fc2_b  = (const float*)d_in[6];
  const float* mlp1_w = (const float*)d_in[7];
  const float* mlp1_b = (const float*)d_in[8];
  const float* mlp2_w = (const float*)d_in[9];
  const float* mlp2_b = (const float*)d_in[10];
  const float* gn1_w  = (const float*)d_in[11];
  const float* gn1_b  = (const float*)d_in[12];
  const float* gn2_w  = (const float*)d_in[13];
  const float* gn2_b  = (const float*)d_in[14];
  const float* gn3_w  = (const float*)d_in[15];
  const float* gn3_b  = (const float*)d_in[16];

  _Float16* wsh = (_Float16*)d_ws;

  cvt_weights<<<(WS_TOT + 255) / 256, 256, 0, stream>>>(
      fc1_w, fc2_w, mlp1_w, mlp2_w, wsh);

  ltae_kernel<<<2048, 256, 0, stream>>>(
      x, doy, pmask, fc1_b, fc2_b, mlp1_b, mlp2_b,
      gn1_w, gn1_b, gn2_w, gn2_b, gn3_w, gn3_b,
      wsh, (float*)d_out);
}
